// HGraphSAGE_64931315581555
// MI455X (gfx1250) — matrix operand order repaired and checked by reading, not compiled, not run on hardware
//
#include <hip/hip_runtime.h>

#define RR 3
#define NN 50000
#define EE 500000
#define IND 256
#define HID 128
#define SEMH 128

typedef __attribute__((ext_vector_type(16))) __bf16          bf16x16;
typedef __attribute__((ext_vector_type(8)))  unsigned int    u32x8;
typedef __attribute__((ext_vector_type(8)))  float           f32x8;
typedef __attribute__((ext_vector_type(4)))  unsigned int    uint32x4;
typedef __attribute__((ext_vector_type(8)))  int             int32x8;
typedef __attribute__((ext_vector_type(4)))  int             int32x4;
typedef unsigned short ushort_t;

#if __has_builtin(__builtin_amdgcn_tensor_load_to_lds)
#define USE_TDM 1
#else
#define USE_TDM 0
#endif

// ---------- helpers ----------
__device__ inline unsigned short f2bf_bits(float f) {
    unsigned u = __float_as_uint(f);
    return (unsigned short)((u + 0x7fffu + ((u >> 16) & 1u)) >> 16); // RNE
}
__device__ inline float bf2f(unsigned short u) {
    return __uint_as_float((unsigned)u << 16);
}
// order-preserving float<->int key for atomicMax over floats
__device__ inline int fkey(float f) {
    int b = __float_as_int(f);
    return b >= 0 ? b : (b ^ 0x7fffffff);
}
__device__ inline float keyf(int k) {
    int b = k >= 0 ? k : (k ^ 0x7fffffff);
    return __int_as_float(b);
}

// ---------- stream convert f32 -> bf16 (4 elems/thread) ----------
__global__ void cvt_kernel(const float* __restrict__ in, ushort_t* __restrict__ out, long n4)
{
    long i = blockIdx.x * 256L + threadIdx.x;
    if (i >= n4) return;
    float4 v = ((const float4*)in)[i];
    ushort4 o = make_ushort4(f2bf_bits(v.x), f2bf_bits(v.y), f2bf_bits(v.z), f2bf_bits(v.w));
    ((ushort4*)out)[i] = o;
}

// ---------- pack weight B[K,128] (f32) into WMMA B-fragment bf16 layout ----------
// Bp[((kb*8 + wv)*32 + lane)*16 + i] = bf16( B[(kb*32 + (i>>3)*16 + (lane>>4)*8 + (i&7))*128 + wv*16 + (lane&15)] )
__global__ void pack_b_kernel(const float* __restrict__ B, ushort_t* __restrict__ Bp, int K)
{
    int b = blockIdx.y;
    B  += (size_t)b * K * HID;
    Bp += (size_t)b * K * HID;
    int j = blockIdx.x * 256 + threadIdx.x;
    if (j >= K * HID) return;
    int i    = j & 15;
    int lane = (j >> 4) & 31;
    int wv   = (j >> 9) & 7;
    int kb   = j >> 12;
    int src  = (kb * 32 + (i >> 3) * 16 + (lane >> 4) * 8 + (i & 7)) * HID + wv * 16 + (lane & 15);
    Bp[j] = f2bf_bits(B[src]);
}

// ---------- WMMA GEMM (bf16 in, f32 acc): C[M,128] = act(A @ B + bias) ----------
// block = 256 threads = 8 waves; wave wv -> 16x16 tile at cols [16wv,16wv+16)
// A tile (16 x KDIM, contiguous) staged into LDS by the Tensor Data Mover (wave 0),
// consumed by all 8 waves via ds_load_b128. B read from global in packed-fragment form.
template <int KDIM, int ACT>
__global__ __launch_bounds__(256) void wmma_gemm_kernel(
    const ushort_t* __restrict__ A, const ushort_t* __restrict__ Bp,
    const float* __restrict__ bias, float* __restrict__ C, ushort_t* __restrict__ Cbf,
    long strideA, long strideB, long strideBias, long strideC)
{
    const int b = blockIdx.y;
    A  += (size_t)b * strideA;
    Bp += (size_t)b * strideB;
    if (bias) bias += (size_t)b * strideBias;
    if (C)    C    += (size_t)b * strideC;
    if (Cbf)  Cbf  += (size_t)b * strideC;

    const int lane = threadIdx.x & 31;
    const int wv   = threadIdx.x >> 5;    // 0..7 -> col block
    const int half = lane >> 4;           // 0/1
    const int l16  = lane & 15;

    const int col = wv * 16 + l16;        // output column

    __shared__ ushort_t atile[16 * KDIM]; // 16 rows x KDIM bf16 (<= 8KB)

#if USE_TDM
    // ---- TDM: one 1-D DMA of the contiguous 16xKDIM A tile into LDS ----
    if ((threadIdx.x >> 5) == 0) {        // wave 0 only (EXEC all-ones in-wave)
        const ushort_t* gsrc = A + (size_t)blockIdx.x * 16 * KDIM;
        unsigned long long ga = (unsigned long long)(const void*)gsrc;
        const unsigned elems = 16u * (unsigned)KDIM;   // in data_size units (2B)
        uint32x4 g0;
        g0[0] = 1u;                                          // count=1, user mode
        g0[1] = 0u;                                          // lds_addr = 0 (atile @ LDS base)
        g0[2] = (unsigned)(ga & 0xffffffffull);              // global_addr[31:0]
        g0[3] = (unsigned)((ga >> 32) & 0x1ffffffull) | (2u << 30); // addr[56:32] | type=2
        int32x8 g1;
        g1[0] = (int)(1u << 16);                             // data_size=1 (2 bytes)
        g1[1] = (int)((elems & 0xffffu) << 16);              // tensor_dim0[15:0] @ bits63:48
        g1[2] = (int)(((elems >> 16) & 0xffffu) | (1u << 16)); // dim0[31:16] | tensor_dim1=1
        g1[3] = (int)((elems & 0xffffu) << 16);              // tile_dim0 @ bits127:112
        g1[4] = 0;                                           // tile_dim1=0, tile_dim2=0
        g1[5] = (int)elems;                                  // tensor_dim0_stride[31:0]
        g1[6] = 0;
        g1[7] = 0;
        int32x4 gz4 = {0, 0, 0, 0};
        int32x8 gz8 = {0, 0, 0, 0, 0, 0, 0, 0};
        __builtin_amdgcn_tensor_load_to_lds(g0, g1, gz4, gz4, gz8, 0);
        __builtin_amdgcn_s_wait_tensorcnt(0);
    }
    __syncthreads();
#else
    // fallback: cooperative vector-load of the A tile into LDS
    for (int j = threadIdx.x * 8; j < 16 * KDIM; j += 256 * 8) {
        const uint4 v = *(const uint4*)(A + (size_t)blockIdx.x * 16 * KDIM + j);
        *(uint4*)(atile + j) = v;
    }
    __syncthreads();
#endif

    const ushort_t* arowl = atile + (unsigned)l16 * KDIM;

    f32x8 acc = {0.f, 0.f, 0.f, 0.f, 0.f, 0.f, 0.f, 0.f};

#pragma unroll
    for (int k0 = 0; k0 < KDIM; k0 += 32) {
        // A fragment from LDS: lane holds K=[k0+half*8 .. +7] and K=[k0+16+half*8 .. +7]
        const uint4* pa = (const uint4*)(arowl + k0 + half * 8);
        uint4 a0 = pa[0];      // 8 bf16
        uint4 a1 = pa[2];      // +16 elements = +2 uint4
        // B fragment: contiguous 32B per lane in packed layout (global, L2-resident)
        const uint4* pb = (const uint4*)(Bp + ((size_t)((k0 >> 5) * 8 + wv) * 32 + lane) * 16);
        uint4 b0 = pb[0];
        uint4 b1 = pb[1];

        u32x8 ap, bp;
        ap[0] = a0.x; ap[1] = a0.y; ap[2] = a0.z; ap[3] = a0.w;
        ap[4] = a1.x; ap[5] = a1.y; ap[6] = a1.z; ap[7] = a1.w;
        bp[0] = b0.x; bp[1] = b0.y; bp[2] = b0.z; bp[3] = b0.w;
        bp[4] = b1.x; bp[5] = b1.y; bp[6] = b1.z; bp[7] = b1.w;

        acc = __builtin_amdgcn_wmma_f32_16x16x32_bf16(
            false, __builtin_bit_cast(bf16x16, ap),
            false, __builtin_bit_cast(bf16x16, bp),
            (short)0, acc, false, false);
    }

    // C layout: lane<16 -> rows v (0..7), lane>=16 -> rows v+8 ; N = l16
    const float bv = bias ? bias[col] : 0.f;
#pragma unroll
    for (int v = 0; v < 8; ++v) {
        int orow = blockIdx.x * 16 + half * 8 + v;
        float val = acc[v] + bv;
        if (ACT == 1) val = tanhf(val);
        if (C)   C[(size_t)orow * HID + col] = val;
        if (Cbf) Cbf[(size_t)orow * HID + col] = f2bf_bits(val);
    }
}

// ---------- init (z accumulators, softmax stats) ----------
__global__ void init_kernel(float* __restrict__ z, float* __restrict__ ssum,
                            int* __restrict__ mkeys, float* __restrict__ wsum)
{
    const long total = (long)RR * NN * HID;
    for (long i = blockIdx.x * (long)blockDim.x + threadIdx.x; i < total;
         i += (long)gridDim.x * blockDim.x) {
        z[i] = 0.f;
        if (i < (long)RR * NN) { ssum[i] = 0.f; mkeys[i] = (int)0x80000000; }
        if (i < RR) wsum[i] = 0.f;
    }
}

// ---------- wal[r] = W_gat[r] @ attn_l[r], war likewise ----------
__global__ void walwar_kernel(const float* __restrict__ Wg, const float* __restrict__ al,
                              const float* __restrict__ ar, float* __restrict__ wal,
                              float* __restrict__ war)
{
    int idx = blockIdx.x * 256 + threadIdx.x;
    if (idx >= RR * HID) return;
    int r = idx / HID, i = idx % HID;
    const float* W = Wg + ((size_t)r * HID + i) * HID;
    const float* alp = al + r * HID;
    const float* arp = ar + r * HID;
    float sl = 0.f, sr = 0.f;
    for (int j = 0; j < HID; ++j) { float w = W[j]; sl += w * alp[j]; sr += w * arp[j]; }
    wal[idx] = sl; war[idx] = sr;
}

// ---------- el[r,n] = h_src_bf[r,n]·wal[r] ; er[r,n] = h_dst[n]·war[r] ----------
__global__ void eler_kernel(const ushort_t* __restrict__ h_src_bf, const float* __restrict__ h_dst,
                            const float* __restrict__ wal, const float* __restrict__ war,
                            float* __restrict__ el, float* __restrict__ er)
{
    int idx = blockIdx.x * 256 + threadIdx.x;
    if (idx >= RR * NN) return;
    int r = idx / NN, n = idx % NN;
    const ushort_t* ap = h_src_bf + (size_t)idx * HID;
    const float* dp = h_dst + (size_t)n * HID;
    const float* lp = wal + r * HID;
    const float* rp = war + r * HID;
    float sl = 0.f, sr = 0.f;
#pragma unroll 8
    for (int j = 0; j < HID; ++j) {
        sl += bf2f(ap[j]) * lp[j];
        sr += dp[j] * rp[j];
    }
    el[idx] = sl; er[idx] = sr;
}

// ---------- edge pass 1: leaky-relu + segment max (ordered-int atomicMax) ----------
__global__ void edge_max_kernel(const int* __restrict__ si, const int* __restrict__ di,
                                const float* __restrict__ el, const float* __restrict__ er,
                                float* __restrict__ ebuf, int* __restrict__ mkeys)
{
    long idx = blockIdx.x * 256L + threadIdx.x;
    if (idx >= (long)RR * EE) return;
    int r = (int)(idx / EE);
    int s = si[idx], d = di[idx];
    float e = el[r * NN + s] + er[r * NN + d];
    e = e > 0.f ? e : 0.2f * e;
    ebuf[idx] = e;
    atomicMax(&mkeys[r * NN + d], fkey(e));
}

// ---------- edge pass 2: exp + segment sum ----------
__global__ void edge_exp_kernel(const int* __restrict__ di, float* __restrict__ ebuf,
                                const int* __restrict__ mkeys, float* __restrict__ ssum)
{
    long idx = blockIdx.x * 256L + threadIdx.x;
    if (idx >= (long)RR * EE) return;
    int r = (int)(idx / EE);
    int d = di[idx];
    float m = keyf(mkeys[r * NN + d]);
    float ex = __expf(ebuf[idx] - m);
    ebuf[idx] = ex;
    atomicAdd(&ssum[r * NN + d], ex);
}

// ---------- edge pass 3: alpha-weighted scatter-add (32 lanes/edge, 4 floats/lane) ----------
__global__ void edge_agg_kernel(const int* __restrict__ si, const int* __restrict__ di,
                                const float* __restrict__ ebuf, const float* __restrict__ ssum,
                                const float* __restrict__ hsb, float* __restrict__ zb)
{
    long t = blockIdx.x * 256L + threadIdx.x;
    long eidx = t >> 5;
    if (eidx >= (long)RR * EE) return;
    int lane = (int)(t & 31);
    int r = (int)(eidx / EE);
    int s = si[eidx], d = di[eidx];
    float alpha = ebuf[eidx] / ssum[r * NN + d];
    const float4 v = ((const float4*)(hsb + ((size_t)r * NN + s) * HID))[lane];
    float* dst = zb + ((size_t)r * NN + d) * HID + lane * 4;
    atomicAdd(dst + 0, alpha * v.x);
    atomicAdd(dst + 1, alpha * v.y);
    atomicAdd(dst + 2, alpha * v.z);
    atomicAdd(dst + 3, alpha * v.w);
}

// ---------- elu(z + b_gat) in place, plus bf16 copy for the semantic GEMM ----------
__global__ void elu_bias_kernel(float* __restrict__ zb, ushort_t* __restrict__ z_bf,
                                const float* __restrict__ b_gat)
{
    long idx = blockIdx.x * 256L + threadIdx.x;
    if (idx >= (long)RR * NN * HID) return;
    int r = (int)(idx / ((long)NN * HID));
    int h = (int)(idx % HID);
    float v = zb[idx] + b_gat[r * HID + h];
    v = v > 0.f ? v : (__expf(v) - 1.f);
    zb[idx] = v;
    z_bf[idx] = f2bf_bits(v);
}

// ---------- w_nr = tanh_row · w2 ; accumulate per-relation sums via LDS ----------
__global__ void semreduce_kernel(const float* __restrict__ tb, const float* __restrict__ w2,
                                 float* __restrict__ wsum)
{
    __shared__ float part[RR];
    if (threadIdx.x < RR) part[threadIdx.x] = 0.f;
    __syncthreads();
    long idx = blockIdx.x * 256L + threadIdx.x;
    if (idx < (long)RR * NN) {
        int r = (int)(idx / NN);
        const float4* t4 = (const float4*)(tb + (size_t)idx * SEMH);
        const float4* w4 = (const float4*)w2;
        float acc = 0.f;
#pragma unroll
        for (int i = 0; i < SEMH / 4; ++i) {
            float4 x = t4[i], y = w4[i];
            acc += x.x * y.x + x.y * y.y + x.z * y.z + x.w * y.w;
        }
        atomicAdd(&part[r], acc);   // ds_add_f32
    }
    __syncthreads();
    if (threadIdx.x < RR) atomicAdd(&wsum[threadIdx.x], part[threadIdx.x]);
}

// ---------- a = softmax(wsum / N) over relations ----------
__global__ void asoftmax_kernel(const float* __restrict__ wsum, float* __restrict__ a)
{
    if (threadIdx.x == 0 && blockIdx.x == 0) {
        float v[RR], m = -1e30f, s = 0.f;
        for (int r = 0; r < RR; ++r) { v[r] = wsum[r] / (float)NN; m = fmaxf(m, v[r]); }
        for (int r = 0; r < RR; ++r) { v[r] = __expf(v[r] - m); s += v[r]; }
        for (int r = 0; r < RR; ++r) a[r] = v[r] / s;
    }
}

// ---------- out[n,h] = sum_r a[r] * z[r,n,h] ----------
__global__ void combine_kernel(const float* __restrict__ zb, const float* __restrict__ a,
                               float* __restrict__ out)
{
    int idx = blockIdx.x * 256 + threadIdx.x;
    if (idx >= NN * HID) return;
    float acc = 0.f;
#pragma unroll
    for (int r = 0; r < RR; ++r) acc += a[r] * zb[(size_t)r * NN * HID + idx];
    out[idx] = acc;
}

// ---------- launch ----------
extern "C" void kernel_launch(void* const* d_in, const int* in_sizes, int n_in,
                              void* d_out, int out_size, void* d_ws, size_t ws_size,
                              hipStream_t stream)
{
    const float* dst_feat  = (const float*)d_in[0];
    const float* src_feats = (const float*)d_in[1];
    const int*   src_idx   = (const int*)d_in[2];
    const int*   dst_idx   = (const int*)d_in[3];
    const float* WT_dst    = (const float*)d_in[4];
    const float* bT_dst    = (const float*)d_in[5];
    const float* WT_src    = (const float*)d_in[6];
    const float* bT_src    = (const float*)d_in[7];
    const float* W_gat     = (const float*)d_in[8];
    const float* attn_l    = (const float*)d_in[9];
    const float* attn_r    = (const float*)d_in[10];
    const float* b_gat     = (const float*)d_in[11];
    const float* W1_sem    = (const float*)d_in[12];
    const float* b1_sem    = (const float*)d_in[13];
    const float* w2_sem    = (const float*)d_in[14];
    float* out = (float*)d_out;

    float* ws = (float*)d_ws;
    size_t off = 0;
    auto alloc = [&](size_t nfloats) { float* p = ws + off; off += nfloats; return p; };

    // regions (sizes in f32 words)
    float*    srcbf_region = alloc((size_t)RR * NN * IND / 2); // srcf_bf (bf16), later reused as tb (f32)
    float*    shared_reg   = alloc((size_t)RR * NN * HID / 2); // dstf_bf early, z_bf late
    float*    h_dst        = alloc((size_t)NN * HID);
    float*    hsrcbf_reg   = alloc((size_t)RR * NN * HID / 2); // h_src bf16
    float*    hsb          = alloc((size_t)RR * NN * HID);
    float*    zb           = alloc((size_t)RR * NN * HID);
    float*    el           = alloc((size_t)RR * NN);
    float*    er           = alloc((size_t)RR * NN);
    float*    ssum         = alloc((size_t)RR * NN);
    int*      mkeys        = (int*)alloc((size_t)RR * NN);
    float*    ebuf         = alloc((size_t)RR * EE);
    float*    wal          = alloc((size_t)RR * HID);
    float*    war          = alloc((size_t)RR * HID);
    // packed bf16 weights
    ushort_t* WTdst_p      = (ushort_t*)alloc((size_t)IND * HID / 2);
    ushort_t* WTsrc_p      = (ushort_t*)alloc((size_t)RR * IND * HID / 2);
    ushort_t* Wgat_p       = (ushort_t*)alloc((size_t)RR * HID * HID / 2);
    ushort_t* W1_p         = (ushort_t*)alloc((size_t)HID * HID / 2);
    float*    wsum         = alloc(RR);
    float*    a            = alloc(RR);

    ushort_t* srcf_bf  = (ushort_t*)srcbf_region;
    float*    tb       = srcbf_region;           // reuse after GEMM1-src
    ushort_t* dstf_bf  = (ushort_t*)shared_reg;  // early
    ushort_t* z_bf     = (ushort_t*)shared_reg;  // late (after dstf_bf dead)
    ushort_t* h_src_bf = (ushort_t*)hsrcbf_reg;
    (void)in_sizes; (void)n_in; (void)out_size; (void)ws_size;

    dim3 blk(256);

    init_kernel<<<4096, blk, 0, stream>>>(zb, ssum, mkeys, wsum);

    // activations -> bf16
    cvt_kernel<<<(unsigned)(((long)NN * IND / 4 + 255) / 256), blk, 0, stream>>>(
        dst_feat, dstf_bf, (long)NN * IND / 4);
    cvt_kernel<<<(unsigned)(((long)RR * NN * IND / 4 + 255) / 256), blk, 0, stream>>>(
        src_feats, srcf_bf, (long)RR * NN * IND / 4);

    // weights -> packed bf16 fragments
    pack_b_kernel<<<dim3((IND * HID + 255) / 256, 1), blk, 0, stream>>>(WT_dst, WTdst_p, IND);
    pack_b_kernel<<<dim3((IND * HID + 255) / 256, RR), blk, 0, stream>>>(WT_src, WTsrc_p, IND);
    pack_b_kernel<<<dim3((HID * HID + 255) / 256, RR), blk, 0, stream>>>(W_gat, Wgat_p, HID);
    pack_b_kernel<<<dim3((HID * HID + 255) / 256, 1), blk, 0, stream>>>(W1_sem, W1_p, HID);

    // h_dst = dst_feat @ WT_dst + bT_dst   (f32 out only)
    wmma_gemm_kernel<IND, 0><<<dim3(NN / 16, 1), blk, 0, stream>>>(
        dstf_bf, WTdst_p, bT_dst, h_dst, nullptr, 0, 0, 0, 0);
    // h_src[r] = src_feats[r] @ WT_src[r] + bT_src[r]   (bf16 out only)
    wmma_gemm_kernel<IND, 0><<<dim3(NN / 16, RR), blk, 0, stream>>>(
        srcf_bf, WTsrc_p, bT_src, nullptr, h_src_bf,
        (long)NN * IND, (long)IND * HID, (long)HID, (long)NN * HID);
    // hs[r] = h_src[r] @ W_gat[r]   (f32 out)
    wmma_gemm_kernel<HID, 0><<<dim3(NN / 16, RR), blk, 0, stream>>>(
        h_src_bf, Wgat_p, nullptr, hsb, nullptr,
        (long)NN * HID, (long)HID * HID, 0, (long)NN * HID);

    walwar_kernel<<<(RR * HID + 255) / 256, blk, 0, stream>>>(W_gat, attn_l, attn_r, wal, war);
    eler_kernel<<<(RR * NN + 255) / 256, blk, 0, stream>>>(h_src_bf, h_dst, wal, war, el, er);

    edge_max_kernel<<<(RR * EE + 255) / 256, blk, 0, stream>>>(src_idx, dst_idx, el, er, ebuf, mkeys);
    edge_exp_kernel<<<(RR * EE + 255) / 256, blk, 0, stream>>>(dst_idx, ebuf, mkeys, ssum);
    long aggT = (long)RR * EE * 32;
    edge_agg_kernel<<<(unsigned)((aggT + 255) / 256), blk, 0, stream>>>(
        src_idx, dst_idx, ebuf, ssum, hsb, zb);
    elu_bias_kernel<<<(unsigned)(((long)RR * NN * HID + 255) / 256), blk, 0, stream>>>(
        zb, z_bf, b_gat);

    // t = tanh(z @ W1_sem + b1_sem), rows = R*N (multiple of 16)
    wmma_gemm_kernel<HID, 1><<<dim3(RR * NN / 16, 1), blk, 0, stream>>>(
        z_bf, W1_p, b1_sem, tb, nullptr, 0, 0, 0, 0);
    semreduce_kernel<<<(RR * NN + 255) / 256, blk, 0, stream>>>(tb, w2_sem, wsum);
    asoftmax_kernel<<<1, 32, 0, stream>>>(wsum, a);
    combine_kernel<<<(NN * HID + 255) / 256, blk, 0, stream>>>(zb, a, out);
}
